// CPMAntAttention_17471926960781
// MI455X (gfx1250) — matrix operand order repaired and checked by reading, not compiled, not run on hardware
//
#include <hip/hip_runtime.h>

// Problem constants (CPMAnt attention): B=2, S=1024, D=4096, H=32, DH=128
#define B_  2
#define S_  1024
#define D_  4096
#define H_  32
#define DH_ 128

typedef __attribute__((ext_vector_type(16))) __bf16 v16bf;
typedef __attribute__((ext_vector_type(8)))  __bf16 v8bf;
typedef __attribute__((ext_vector_type(8)))  float  v8f;

__device__ __forceinline__ v16bf cat16(v8bf lo, v8bf hi) {
  return __builtin_shufflevector(lo, hi, 0,1,2,3,4,5,6,7,8,9,10,11,12,13,14,15);
}

// D(16x16,f32) = A(16x32,bf16) * B(32x16,bf16) + C
__device__ __forceinline__ v8f wmma_bf16(v16bf a, v16bf b, v8f c) {
  return __builtin_amdgcn_wmma_f32_16x16x32_bf16(false, a, false, b,
                                                 (short)0, c, false, false);
}

// ---------------------------------------------------------------------------
// Kernel 1: QKV projections.  grid.z selects Q/K/V.
//   X (2048 x 4096 f32) * W (4096 x 4096 f32)  ->  (b,h,s,dh) bf16
// Block tile 128x128, K-step 32, software-pipelined double-buffered LDS:
//   per k-step: issue next tile's global loads -> 8 WMMA on current buffer
//   -> cvt+ds_store next buffer -> one barrier.
// ---------------------------------------------------------------------------
__global__ void __launch_bounds__(256)
qkv_gemm_kernel(const float* __restrict__ Xq, const float* __restrict__ Xkv,
                const float* __restrict__ Wq, const float* __restrict__ Wk,
                const float* __restrict__ Wv,
                __bf16* __restrict__ Qws, __bf16* __restrict__ Kws,
                __bf16* __restrict__ Vws)
{
  const int z = blockIdx.z;
  const float* __restrict__ X = (z == 0) ? Xq : Xkv;
  const float* __restrict__ W = (z == 0) ? Wq : ((z == 1) ? Wk : Wv);
  __bf16* __restrict__ Out = (z == 0) ? Qws : ((z == 1) ? Kws : Vws);

  __shared__ __align__(32) __bf16 As[2][128][32];   // 16 KB
  __shared__ __align__(32) __bf16 Bs[2][32][128];   // 16 KB

  const int tid  = threadIdx.x;
  const int lane = tid & 31;
  const int wave = tid >> 5;   // 0..7
  const int wm   = wave >> 1;  // 0..3
  const int wn   = wave & 1;   // 0..1
  const int hf   = lane >> 4;  // lane half (wave32 fragment layout)
  const int l16  = lane & 15;

  const int m0 = blockIdx.y * 128;
  const int n0 = blockIdx.x * 128;

  v8f acc[2][4];
#pragma unroll
  for (int i = 0; i < 2; ++i)
#pragma unroll
    for (int j = 0; j < 4; ++j) acc[i][j] = v8f{};

  const int arow = tid >> 1, acol = (tid & 1) * 16;  // 128x32 / 256 thr
  const int brow = tid >> 3, bcol = (tid & 7) * 16;  // 32x128 / 256 thr

  float areg[16], breg[16];
  auto load_regs = [&](int k0) {
    const float* ap = X + (size_t)(m0 + arow) * D_ + k0 + acol;
    const float* bp = W + (size_t)(k0 + brow) * D_ + n0 + bcol;
#pragma unroll
    for (int j = 0; j < 16; ++j) { areg[j] = ap[j]; breg[j] = bp[j]; }
  };
  auto store_lds = [&](int buf) {
#pragma unroll
    for (int j = 0; j < 16; ++j) {
      As[buf][arow][acol + j] = (__bf16)areg[j];
      Bs[buf][brow][bcol + j] = (__bf16)breg[j];
    }
  };

  const int NK = D_ / 32;
  load_regs(0);
  store_lds(0);
  __syncthreads();

  for (int i = 0; i < NK; ++i) {
    const int cur = i & 1;
    if (i + 1 < NK) load_regs((i + 1) * 32);           // overlap HBM with WMMA
    if (i + 2 < NK) {                                  // global_prefetch_b8
      __builtin_prefetch(X + (size_t)(m0 + arow) * D_ + (i + 2) * 32 + acol, 0, 1);
      __builtin_prefetch(W + (size_t)((i + 2) * 32 + brow) * D_ + n0 + bcol, 0, 1);
    }

    // B fragments: column = l16, elements over K in table order
    v16bf bf[4];
#pragma unroll
    for (int nt = 0; nt < 4; ++nt)
      for (int s = 0; s < 16; ++s) bf[nt][s] = Bs[cur][(s & 8) * 2 + hf * 8 + (s & 7)][wn * 64 + nt * 16 + l16];
#pragma unroll
    for (int mt = 0; mt < 2; ++mt) {
      // A fragment: lane l16 = row; half picks K chunks {h*8.., 16+h*8..}
      const int r = wm * 32 + mt * 16 + l16;
      v8bf lo = *(const v8bf*)&As[cur][r][hf * 8];
      v8bf hi = *(const v8bf*)&As[cur][r][16 + hf * 8];
      v16bf af = cat16(lo, hi);
#pragma unroll
      for (int nt = 0; nt < 4; ++nt)
        acc[mt][nt] = wmma_bf16(af, bf[nt], acc[mt][nt]);
    }

    if (i + 1 < NK) store_lds(cur ^ 1);
    __syncthreads();   // cur-buffer readers done before it is rewritten
  }

  // C layout: VGPR r, lane -> (M = r + 8*half, N = l16). Scatter to (b,h,s,dh).
#pragma unroll
  for (int mt = 0; mt < 2; ++mt)
#pragma unroll
    for (int nt = 0; nt < 4; ++nt)
#pragma unroll
      for (int r = 0; r < 8; ++r) {
        const int grow = m0 + wm * 32 + mt * 16 + r + 8 * hf; // b*S + s
        const int gcol = n0 + wn * 64 + nt * 16 + l16;        // h*DH + dh
        const int b = grow >> 10, s = grow & (S_ - 1);
        const int h = gcol >> 7,  dh = gcol & (DH_ - 1);
        Out[(((size_t)b * H_ + h) * S_ + s) * DH_ + dh] = (__bf16)acc[mt][nt][r];
      }
}

// ---------------------------------------------------------------------------
// Kernel 2: flash attention per (b,h).  4 waves/block, each wave owns a
// 16-row Q tile; 32-key blocks double-buffered in LDS (K transposed so score
// B-fragments are contiguous rows).  One barrier per key-block; the P-tile
// C->A relayout uses wave-private LDS (same-wave DS ops are in-order).
// ---------------------------------------------------------------------------
__global__ void __launch_bounds__(128)
attn_kernel(const __bf16* __restrict__ Q, const __bf16* __restrict__ K,
            const __bf16* __restrict__ V, const float* __restrict__ pb,
            const unsigned char* __restrict__ msk, __bf16* __restrict__ Ctx)
{
  __shared__ __align__(32) __bf16 Kt[2][128][32];   // K^T: [d][key]  16 KB
  __shared__ __align__(32) __bf16 Vs[2][32][128];   // [key][dh]      16 KB
  __shared__ __align__(32) __bf16 Pw[4][16][32];    // per-wave P      4 KB

  const int tid  = threadIdx.x;
  const int lane = tid & 31;
  const int wave = tid >> 5;   // 0..3
  const int hf   = lane >> 4;
  const int l16  = lane & 15;

  const int b  = blockIdx.y >> 5;       // H_ == 32
  const int h  = blockIdx.y & (H_ - 1);
  const int q0 = blockIdx.x * 64 + wave * 16;

  const size_t head = ((size_t)b * H_ + h) * S_;

  // Preload Q fragments: 16 rows x 128 d, 4 K-chunks of 32, A layout.
  v16bf qf[4];
  {
    const __bf16* qb = Q + (head + q0 + l16) * DH_;
#pragma unroll
    for (int kk = 0; kk < 4; ++kk) {
      v8bf lo = *(const v8bf*)(qb + kk * 32 + hf * 8);
      v8bf hi = *(const v8bf*)(qb + kk * 32 + 16 + hf * 8);
      qf[kk] = cat16(lo, hi);
    }
  }

  v8f acc[8];
#pragma unroll
  for (int i = 0; i < 8; ++i) acc[i] = v8f{};
  float mrow[8], lrow[8];
#pragma unroll
  for (int r = 0; r < 8; ++r) { mrow[r] = -1e30f; lrow[r] = 0.0f; }

  const float sc = 0.08838834764831845f;  // 1/sqrt(DH)

  const int kkey  = tid >> 2;        // 0..31 (key row this thread stages)
  const int dpart = (tid & 3) * 32;  // 32 d's per thread

  __bf16 kreg[32], vreg[32];
  auto load_kv_regs = [&](int k0) {
    const __bf16* kp = K + (head + k0 + kkey) * DH_ + dpart;
    const __bf16* vp = V + (head + k0 + kkey) * DH_ + dpart;
#pragma unroll
    for (int j = 0; j < 4; ++j) {
      *(v8bf*)&kreg[j * 8] = *(const v8bf*)(kp + j * 8);
      *(v8bf*)&vreg[j * 8] = *(const v8bf*)(vp + j * 8);
    }
  };
  auto store_kv_lds = [&](int buf) {
#pragma unroll
    for (int j = 0; j < 32; ++j) Kt[buf][dpart + j][kkey] = kreg[j];  // transpose
#pragma unroll
    for (int j = 0; j < 4; ++j)
      *(v8bf*)&Vs[buf][kkey][dpart + j * 8] = *(const v8bf*)&vreg[j * 8];
  };

  const int NB = S_ / 32;
  load_kv_regs(0);
  store_kv_lds(0);
  __syncthreads();

  for (int kb = 0; kb < NB; ++kb) {
    const int cur = kb & 1;
    const int k0  = kb * 32;
    if (kb + 1 < NB) {
      load_kv_regs(k0 + 32);                       // overlap HBM with compute
#pragma unroll
      for (int r = 0; r < 8; ++r)                  // prefetch next bias rows
        __builtin_prefetch(pb + (head + q0 + r + 8 * hf) * S_ + k0 + 32 + l16, 0, 1);
    }

    // S = Q * K^T  (two 16-key tiles, contraction over DH in 4 chunks)
    v8f st0 = v8f{}, st1 = v8f{};
#pragma unroll
    for (int kk = 0; kk < 4; ++kk) {
      v16bf b0, b1;
      for (int s = 0; s < 16; ++s) {
        const int d = kk * 32 + (s & 8) * 2 + hf * 8 + (s & 7);
        b0[s] = Kt[cur][d][l16];
        b1[s] = Kt[cur][d][16 + l16];
      }
      st0 = wmma_bf16(qf[kk], b0, st0);
      st1 = wmma_bf16(qf[kk], b1, st1);
    }

    // scale + position bias + mask (C layout element = (q0+r+8*hf, k0+..+l16))
    unsigned mb0 = 0u, mb1 = 0u;
#pragma unroll
    for (int r = 0; r < 8; ++r) {
      const int q = q0 + r + 8 * hf;
      const size_t poff = (head + q) * S_;
      const size_t moff = ((size_t)b * S_ + q) * S_;
      {
        const int k = k0 + l16;
        float s = st0[r] * sc + pb[poff + k];
        bool keep = msk[moff + k] != 0;
        st0[r] = keep ? s : -1e30f;
        mb0 |= keep ? (1u << r) : 0u;
      }
      {
        const int k = k0 + 16 + l16;
        float s = st1[r] * sc + pb[poff + k];
        bool keep = msk[moff + k] != 0;
        st1[r] = keep ? s : -1e30f;
        mb1 |= keep ? (1u << r) : 0u;
      }
    }

    // online softmax: rows live in one 16-lane half -> xor-shuffle reduce
    float rmax[8];
#pragma unroll
    for (int r = 0; r < 8; ++r) rmax[r] = fmaxf(st0[r], st1[r]);
#pragma unroll
    for (int off = 1; off < 16; off <<= 1)
#pragma unroll
      for (int r = 0; r < 8; ++r)
        rmax[r] = fmaxf(rmax[r], __shfl_xor(rmax[r], off, 32));

    float scl[8];
#pragma unroll
    for (int r = 0; r < 8; ++r) {
      float mn = fmaxf(mrow[r], rmax[r]);
      scl[r] = __expf(mrow[r] - mn);
      mrow[r] = mn;
    }

    float ps[8];
#pragma unroll
    for (int r = 0; r < 8; ++r) {
      float p0 = ((mb0 >> r) & 1u) ? __expf(st0[r] - mrow[r]) : 0.0f;
      float p1 = ((mb1 >> r) & 1u) ? __expf(st1[r] - mrow[r]) : 0.0f;
      st0[r] = p0; st1[r] = p1;
      ps[r] = p0 + p1;
    }
#pragma unroll
    for (int off = 1; off < 16; off <<= 1)
#pragma unroll
      for (int r = 0; r < 8; ++r)
        ps[r] += __shfl_xor(ps[r], off, 32);
#pragma unroll
    for (int r = 0; r < 8; ++r) lrow[r] = lrow[r] * scl[r] + ps[r];

#pragma unroll
    for (int nt = 0; nt < 8; ++nt)
#pragma unroll
      for (int r = 0; r < 8; ++r) acc[nt][r] *= scl[r];

    // C-layout P -> wave-private LDS -> A-layout fragment.
    // Same-wave DS ops execute in order: no block barrier needed.
#pragma unroll
    for (int r = 0; r < 8; ++r) {
      Pw[wave][r + 8 * hf][l16]      = (__bf16)st0[r];
      Pw[wave][r + 8 * hf][16 + l16] = (__bf16)st1[r];
    }
    v16bf pf;
    {
      v8bf lo = *(const v8bf*)&Pw[wave][l16][hf * 8];
      v8bf hi = *(const v8bf*)&Pw[wave][l16][16 + hf * 8];
      pf = cat16(lo, hi);
    }
    // ctx += P * V   (B fragment: column dh = l16, elements over key in table order)
#pragma unroll
    for (int nt = 0; nt < 8; ++nt) {
      v16bf vf;
      for (int s = 0; s < 16; ++s) vf[s] = Vs[cur][(s & 8) * 2 + hf * 8 + (s & 7)][nt * 16 + l16];
      acc[nt] = wmma_bf16(pf, vf, acc[nt]);
    }

    if (kb + 1 < NB) store_kv_lds(cur ^ 1);
    __syncthreads();   // cur-buffer readers done before it is rewritten
  }

  float inv[8];
#pragma unroll
  for (int r = 0; r < 8; ++r)
    inv[r] = (lrow[r] > 0.0f) ? (1.0f / lrow[r]) : 0.0f;  // fully-masked row -> 0

  // store ctx as (b, s, h*dh) bf16 for the output GEMM
#pragma unroll
  for (int nt = 0; nt < 8; ++nt)
#pragma unroll
    for (int r = 0; r < 8; ++r) {
      const int q  = q0 + r + 8 * hf;
      const int dh = nt * 16 + l16;
      Ctx[(((size_t)b * S_ + q) * H_ + h) * DH_ + dh] = (__bf16)(acc[nt][r] * inv[r]);
    }
}

// ---------------------------------------------------------------------------
// Kernel 3: out = ctx (2048 x 4096 bf16) * Wo (4096 x 4096 f32->bf16), f32 out
// Same pipelined mainloop as kernel 1.
// ---------------------------------------------------------------------------
__global__ void __launch_bounds__(256)
out_gemm_kernel(const __bf16* __restrict__ Cx, const float* __restrict__ Wo,
                float* __restrict__ Outp)
{
  __shared__ __align__(32) __bf16 As[2][128][32];
  __shared__ __align__(32) __bf16 Bs[2][32][128];

  const int tid  = threadIdx.x;
  const int lane = tid & 31;
  const int wave = tid >> 5;
  const int wm   = wave >> 1;
  const int wn   = wave & 1;
  const int hf   = lane >> 4;
  const int l16  = lane & 15;

  const int m0 = blockIdx.y * 128;
  const int n0 = blockIdx.x * 128;

  v8f acc[2][4];
#pragma unroll
  for (int i = 0; i < 2; ++i)
#pragma unroll
    for (int j = 0; j < 4; ++j) acc[i][j] = v8f{};

  const int arow = tid >> 1, acol = (tid & 1) * 16;
  const int brow = tid >> 3, bcol = (tid & 7) * 16;

  __bf16 areg[16];
  float  breg[16];
  auto load_regs = [&](int k0) {
    const __bf16* ap = Cx + (size_t)(m0 + arow) * D_ + k0 + acol;
    const float*  bp = Wo + (size_t)(k0 + brow) * D_ + n0 + bcol;
    *(v8bf*)&areg[0] = *(const v8bf*)(ap);
    *(v8bf*)&areg[8] = *(const v8bf*)(ap + 8);
#pragma unroll
    for (int j = 0; j < 16; ++j) breg[j] = bp[j];
  };
  auto store_lds = [&](int buf) {
    *(v8bf*)&As[buf][arow][acol]     = *(const v8bf*)&areg[0];
    *(v8bf*)&As[buf][arow][acol + 8] = *(const v8bf*)&areg[8];
#pragma unroll
    for (int j = 0; j < 16; ++j) Bs[buf][brow][bcol + j] = (__bf16)breg[j];
  };

  const int NK = D_ / 32;
  load_regs(0);
  store_lds(0);
  __syncthreads();

  for (int i = 0; i < NK; ++i) {
    const int cur = i & 1;
    if (i + 1 < NK) load_regs((i + 1) * 32);
    if (i + 2 < NK) {
      __builtin_prefetch(Cx + (size_t)(m0 + arow) * D_ + (i + 2) * 32 + acol, 0, 1);
      __builtin_prefetch(Wo + (size_t)((i + 2) * 32 + brow) * D_ + n0 + bcol, 0, 1);
    }

    v16bf bf[4];
#pragma unroll
    for (int nt = 0; nt < 4; ++nt)
      for (int s = 0; s < 16; ++s) bf[nt][s] = Bs[cur][(s & 8) * 2 + hf * 8 + (s & 7)][wn * 64 + nt * 16 + l16];
#pragma unroll
    for (int mt = 0; mt < 2; ++mt) {
      const int r = wm * 32 + mt * 16 + l16;
      v8bf lo = *(const v8bf*)&As[cur][r][hf * 8];
      v8bf hi = *(const v8bf*)&As[cur][r][16 + hf * 8];
      v16bf af = cat16(lo, hi);
#pragma unroll
      for (int nt = 0; nt < 4; ++nt)
        acc[mt][nt] = wmma_bf16(af, bf[nt], acc[mt][nt]);
    }

    if (i + 1 < NK) store_lds(cur ^ 1);
    __syncthreads();
  }

#pragma unroll
  for (int mt = 0; mt < 2; ++mt)
#pragma unroll
    for (int nt = 0; nt < 4; ++nt)
#pragma unroll
      for (int r = 0; r < 8; ++r) {
        const int grow = m0 + wm * 32 + mt * 16 + r + 8 * hf;
        const int gcol = n0 + wn * 64 + nt * 16 + l16;
        Outp[(size_t)grow * D_ + gcol] = acc[mt][nt][r];
      }
}

// ---------------------------------------------------------------------------
extern "C" void kernel_launch(void* const* d_in, const int* in_sizes, int n_in,
                              void* d_out, int out_size, void* d_ws, size_t ws_size,
                              hipStream_t stream) {
  (void)in_sizes; (void)n_in; (void)out_size; (void)ws_size;

  const float*         hq   = (const float*)d_in[0];         // (B,S,D) f32
  const float*         hkv  = (const float*)d_in[1];         // (B,S,D) f32
  const unsigned char* mask = (const unsigned char*)d_in[2]; // (B,S,S) bool
  const float*         pb   = (const float*)d_in[3];         // (B,H,S,S) f32
  const float*         Wq   = (const float*)d_in[4];
  const float*         Wk   = (const float*)d_in[5];
  const float*         Wv   = (const float*)d_in[6];
  const float*         Wo   = (const float*)d_in[7];
  float*               out  = (float*)d_out;                 // (B,S,D) f32

  // workspace: Q | K | V | ctx, each B*H*S*DH bf16 = 16 MB (64 MB total)
  char* wsb = (char*)d_ws;
  const size_t seg = (size_t)B_ * H_ * S_ * DH_ * 2;
  __bf16* Qws = (__bf16*)(wsb);
  __bf16* Kws = (__bf16*)(wsb + seg);
  __bf16* Vws = (__bf16*)(wsb + 2 * seg);
  __bf16* Cws = (__bf16*)(wsb + 3 * seg);

  // 1) QKV projections: (2048x4096)x(4096x4096), z = Q/K/V
  dim3 g1(D_ / 128, (B_ * S_) / 128, 3);
  qkv_gemm_kernel<<<g1, 256, 0, stream>>>(hq, hkv, Wq, Wk, Wv, Qws, Kws, Vws);

  // 2) flash attention per (b,h), 64 q-rows per block
  dim3 g2(S_ / 64, B_ * H_);
  attn_kernel<<<g2, 128, 0, stream>>>(Qws, Kws, Vws, pb, mask, Cws);

  // 3) output projection -> f32
  dim3 g3(D_ / 128, (B_ * S_) / 128);
  out_gemm_kernel<<<g3, 256, 0, stream>>>(Cws, Wo, out);
}
